// Transformer_23261542875269
// MI455X (gfx1250) — matrix operand order repaired and checked by reading, not compiled, not run on hardware
//
#include <hip/hip_runtime.h>
#include <hip/hip_bf16.h>
#include <math.h>

typedef __attribute__((ext_vector_type(16))) _Float16 v16h;
typedef __attribute__((ext_vector_type(8)))  float    v8f;
typedef __attribute__((ext_vector_type(4)))  _Float16 v4h;

#define B_    16
#define C_    128
#define T_    294
#define V_    25
#define M_    117600      // B*T*V tokens
#define INNER 512
#define NT_   42          // T/7
#define MK_   16800       // B*NT*V kv tokens (global attn)
#define KKV   896         // C*7 contraction for global kv proj

static __device__ __forceinline__ v8f wmma_f16(const v16h& a, const v16h& b, const v8f& c) {
  return __builtin_amdgcn_wmma_f32_16x16x32_f16(false, a, false, b, (short)0, c, false, false);
}

// ---------------------------------------------------------------------------
// Layout transforms
// ---------------------------------------------------------------------------
__global__ __launch_bounds__(256) void to_tokens(const float* __restrict__ x,
                                                 float* __restrict__ X) {
  size_t id = (size_t)blockIdx.x * 256 + threadIdx.x;   // over M_*128
  if (id >= (size_t)M_ * 128) return;
  int c = (int)(id & 127);
  size_t m = id >> 7;
  int v = (int)(m % V_);
  size_t mt = m / V_;
  int t = (int)(mt % T_);
  int b = (int)(mt / T_);
  X[id] = x[(((size_t)b * C_ + c) * T_ + t) * V_ + v];
}

__global__ __launch_bounds__(256) void from_tokens(const float* __restrict__ X,
                                                   float* __restrict__ out) {
  size_t id = (size_t)blockIdx.x * 256 + threadIdx.x;   // over B*C*T*V
  if (id >= (size_t)M_ * 128) return;
  int v = (int)(id % V_);
  size_t r = id / V_;
  int t = (int)(r % T_);
  r /= T_;
  int c = (int)(r % C_);
  int b = (int)(r / C_);
  out[id] = X[(((size_t)b * T_ + t) * V_ + v) * 128 + c];
}

// ---------------------------------------------------------------------------
// Channel LayerNorm over 128 channels, token-major rows -> f16
// ---------------------------------------------------------------------------
__global__ __launch_bounds__(256) void ln_kernel(const float* __restrict__ X,
                                                 const float* __restrict__ g,
                                                 const float* __restrict__ bta,
                                                 _Float16* __restrict__ out,
                                                 int M) {
  int row = blockIdx.x * 8 + (threadIdx.x >> 5);
  int lane = threadIdx.x & 31;
  if (row >= M) return;
  const float4 xv = *(const float4*)(X + (size_t)row * 128 + lane * 4);
  float s  = xv.x + xv.y + xv.z + xv.w;
  float sq = xv.x * xv.x + xv.y * xv.y + xv.z * xv.z + xv.w * xv.w;
  #pragma unroll
  for (int m = 16; m > 0; m >>= 1) {
    s  += __shfl_xor(s, m, 32);
    sq += __shfl_xor(sq, m, 32);
  }
  float mean = s * (1.0f / 128.0f);
  float var  = sq * (1.0f / 128.0f) - mean * mean;
  float inv  = rsqrtf(var + 1e-5f);
  int c = lane * 4;
  v4h o;
  o[0] = (_Float16)((xv.x - mean) * inv * g[c + 0] + bta[c + 0]);
  o[1] = (_Float16)((xv.y - mean) * inv * g[c + 1] + bta[c + 1]);
  o[2] = (_Float16)((xv.z - mean) * inv * g[c + 2] + bta[c + 2]);
  o[3] = (_Float16)((xv.w - mean) * inv * g[c + 3] + bta[c + 3]);
  *(v4h*)(out + (size_t)row * 128 + c) = o;
}

// ---------------------------------------------------------------------------
// WMMA GEMM: C[M,N] = A[M,K](f16) x B[K,N](f32 weights, converted to f16)
// block tile 128x128, 8 waves (4Mx2N), each wave 32x64 = 8 accumulators
// EPI: 0=store f16, 2=bias+GELU store f16, 3=bias+residual(f32 X, in place)
// ---------------------------------------------------------------------------
template <int EPI>
__global__ __launch_bounds__(256)
void gemm_wmma(const _Float16* __restrict__ A, const float* __restrict__ Bw,
               const float* __restrict__ bias, _Float16* __restrict__ outH,
               float* __restrict__ resid, int M, int K, int N) {
  __shared__ _Float16 sA[128][40];
  __shared__ _Float16 sB[32][136];
  const int tid  = threadIdx.x;
  const int lane = tid & 31;
  const int w    = tid >> 5;
  const int m0   = blockIdx.y * 128;
  const int n0   = blockIdx.x * 128;
  const int wm   = (w & 3) * 32;     // wave M offset within tile (2 frags)
  const int wn   = (w >> 2) * 64;    // wave N offset within tile (4 frags)

  v8f acc[2][4];
  #pragma unroll
  for (int mi = 0; mi < 2; ++mi)
    #pragma unroll
    for (int s = 0; s < 4; ++s) acc[mi][s] = v8f{};

  const int arow = tid >> 2;
  const int acg  = (tid & 3) * 8;

  for (int k0 = 0; k0 < K; k0 += 32) {
    // --- stage A tile (128x32 f16), M-guarded; prefetch next k-tile ---
    #pragma unroll
    for (int it = 0; it < 2; ++it) {
      int rr = arow + it * 64;
      int gm = m0 + rr;
      float4 av;
      if (gm < M) {
        av = *(const float4*)(A + (size_t)gm * K + k0 + acg);
        if (k0 + 32 < K)
          __builtin_prefetch((const void*)(A + (size_t)gm * K + k0 + 32 + acg), 0, 1);
      } else {
        av = float4{0.f, 0.f, 0.f, 0.f};
      }
      *(float4*)(&sA[rr][acg]) = av;
    }
    // --- stage B tile (32x128), f32 -> f16 convert ---
    #pragma unroll
    for (int it = 0; it < 8; ++it) {
      int idx = tid + it * 256;       // float2 index among 2048
      int p   = idx * 2;
      int kr  = p >> 7;
      int nc  = p & 127;
      float2 bv = *(const float2*)(Bw + (size_t)(k0 + kr) * N + n0 + nc);
      sB[kr][nc]     = (_Float16)bv.x;
      sB[kr][nc + 1] = (_Float16)bv.y;
    }
    __syncthreads();

    // --- A fragments (16x32 f16 per ISA layout) ---
    v16h af[2];
    #pragma unroll
    for (int mi = 0; mi < 2; ++mi) {
      int am = wm + mi * 16 + (lane & 15);
      int kb = (lane >> 4) * 8;
      #pragma unroll
      for (int e = 0; e < 8; ++e) af[mi][e] = sA[am][kb + e];
      #pragma unroll
      for (int e = 0; e < 8; ++e) af[mi][8 + e] = sA[am][kb + 16 + e];
    }
    // --- 4 B fragments (32x16), lane&15 = N, elements = K (table order) ---
    #pragma unroll
    for (int s = 0; s < 4; ++s) {
      v16h bf;
      int nb = wn + s * 16;
      #pragma unroll
      for (int e = 0; e < 16; ++e) bf[e] = sB[((lane >> 4) << 3) + e + (e & 8)][nb + (lane & 15)];
      acc[0][s] = wmma_f16(af[0], bf, acc[0][s]);
      acc[1][s] = wmma_f16(af[1], bf, acc[1][s]);
    }
    __syncthreads();
  }

  // --- epilogue: C/D layout vgpr r -> M=r(+8 for hi lanes), lane&15 -> N ---
  const int cnBase = n0 + wn + (lane & 15);
  #pragma unroll
  for (int mi = 0; mi < 2; ++mi) {
    int cmBase = wm + mi * 16 + ((lane >> 4) << 3);
    #pragma unroll
    for (int s = 0; s < 4; ++s) {
      int nn = cnBase + s * 16;
      #pragma unroll
      for (int r = 0; r < 8; ++r) {
        int gm = m0 + cmBase + r;
        if (gm < M) {
          float val = acc[mi][s][r];
          if (EPI >= 2) val += bias[nn];
          if (EPI == 2) val = 0.5f * val * (1.0f + erff(val * 0.70710678118f));
          if (EPI == 3) {
            float* rp = resid + (size_t)gm * N + nn;
            *rp = val + *rp;
          } else {
            outH[(size_t)gm * N + nn] = (_Float16)val;
          }
        }
      }
    }
  }
}

// ---------------------------------------------------------------------------
// Gather A-operand for the global kv projection:
// AKV[kvtok, c*7+k] = LN(x)[token(b, tk*7+k, v), c]
// ---------------------------------------------------------------------------
__global__ __launch_bounds__(256) void akv_gather(const _Float16* __restrict__ lnb,
                                                  _Float16* __restrict__ AKV) {
  size_t id = (size_t)blockIdx.x * 256 + threadIdx.x;  // over MK_*KKV
  if (id >= (size_t)MK_ * KKV) return;
  int col = (int)(id % KKV);
  size_t kt = id / KKV;
  int c  = col / 7;
  int kk = col % 7;
  int v  = (int)(kt % V_);
  size_t r = kt / V_;
  int tk = (int)(r % NT_);
  int b  = (int)(r / NT_);
  size_t tok = ((size_t)b * T_ + tk * 7 + kk) * V_ + v;
  AKV[id] = lnb[tok * 128 + c];
}

// ---------------------------------------------------------------------------
// Local attention via WMMA: one wave per (b,nt,v,h) window-head.
// S = Q K^T  : M=7(pad16), N=7(pad16), K=64 -> 2 WMMAs (K^T loaded per-lane
//              straight from global: lane = j, elements = d; zero waste).
// softmax    : half-wave shuffle reductions on accumulator layout, cols>=7
//              masked; padded rows are harmless (never stored).
// O = P V    : P staged per-wave to LDS in A-frag order (cols 7..31 zero),
//              V as B-frag (lane = d, elements = j), 4 N-tiles x 1 k-step.
// ---------------------------------------------------------------------------
__global__ __launch_bounds__(256) void local_attn_wmma(const _Float16* __restrict__ Q,
                                                       const _Float16* __restrict__ KV,
                                                       _Float16* __restrict__ OUT) {
  __shared__ _Float16 sP[8][16][40];   // per-wave P tile, cols 0..31 used
  int tid  = threadIdx.x;
  int lane = tid & 31;
  int w    = tid >> 5;
  int id = blockIdx.x * 8 + w;         // window-head id, 0..134399
  int h  = id & 7;
  int g  = id >> 3;
  int v  = g % V_;
  int g2 = g / V_;
  int nt = g2 % NT_;
  int b  = g2 / NT_;
  int tbase = nt * 7;

  int i  = lane & 15;                  // query row within tile
  int kb = (lane >> 4) * 8;
  bool iOK = (i < 7);
  size_t tokI = ((size_t)b * T_ + tbase + (iOK ? i : 0)) * V_ + v;

  // Q A-fragments for two 32-wide k-steps over d
  v16h qf[2];
  #pragma unroll
  for (int ks = 0; ks < 2; ++ks) {
    if (iOK) {
      const _Float16* qp = Q + tokI * INNER + h * 64 + ks * 32 + kb;
      #pragma unroll
      for (int e = 0; e < 8; ++e) qf[ks][e] = qp[e];
      #pragma unroll
      for (int e = 0; e < 8; ++e) qf[ks][8 + e] = qp[16 + e];
    } else {
      #pragma unroll
      for (int e = 0; e < 16; ++e) qf[ks][e] = (_Float16)0.f;
    }
  }

  // S = Q K^T : single 16-col N tile, 2 k-steps; B-frag lane&15 = j, elem = d
  v8f sacc = v8f{};
  #pragma unroll
  for (int ks = 0; ks < 2; ++ks) {
    int jj = lane & 15;
    v16h bf;
    #pragma unroll
    for (int e = 0; e < 16; ++e) {
      _Float16 val = (_Float16)0.f;
      if (jj < 7) {
        size_t tokJ = ((size_t)b * T_ + tbase + jj) * V_ + v;
        val = KV[tokJ * 1024 + h * 64 + ks * 32 + kb + e + (e & 8)];
      }
      bf[e] = val;
    }
    sacc = wmma_f16(qf[ks], bf, sacc);
  }

  // row softmax over 7 kv columns (col = lane&15, rows rr(+8))
  int rowBase = (lane >> 4) << 3;
  int col = lane & 15;
  bool colOK = (col < 7);
  #pragma unroll
  for (int rr = 0; rr < 8; ++rr) {
    float vv = colOK ? sacc[rr] * 0.125f : -1.0e30f;
    float mx = vv;
    #pragma unroll
    for (int mk = 8; mk > 0; mk >>= 1) mx = fmaxf(mx, __shfl_xor(mx, mk, 32));
    float e0 = __expf(vv - mx);
    float ssum = e0;
    #pragma unroll
    for (int mk = 8; mk > 0; mk >>= 1) ssum += __shfl_xor(ssum, mk, 32);
    float p = e0 / ssum;
    int row = rowBase + rr;
    sP[w][row][col]      = (_Float16)p;
    sP[w][row][col + 16] = (_Float16)0.f;
  }

  // P A-fragment (contraction j padded to 32, single k-step)
  v16h pf;
  {
    int pm = lane & 15;
    #pragma unroll
    for (int e = 0; e < 8; ++e) pf[e] = sP[w][pm][kb + e];
    #pragma unroll
    for (int e = 0; e < 8; ++e) pf[8 + e] = sP[w][pm][kb + 16 + e];
  }

  // O = P V : 4 N-tiles over d; B-frag lane&15 = d, elements = j (table order)
  #pragma unroll
  for (int ntile = 0; ntile < 4; ++ntile) {
    v16h bf;
    #pragma unroll
    for (int e = 0; e < 16; ++e) {
      int j = kb + e + (e & 8);
      _Float16 val = (_Float16)0.f;
      if (j < 7) val = KV[(((size_t)b * T_ + tbase + j) * V_ + v) * 1024 + 512 + h * 64 + ntile * 16 + col];
      bf[e] = val;
    }
    v8f oacc = wmma_f16(pf, bf, v8f{});
    #pragma unroll
    for (int rr = 0; rr < 8; ++rr) {
      int gi = rowBase + rr;
      if (gi < 7) {
        size_t tokO = ((size_t)b * T_ + tbase + gi) * V_ + v;
        OUT[tokO * INNER + h * 64 + ntile * 16 + col] = (_Float16)oacc[rr];
      }
    }
  }
}

// ---------------------------------------------------------------------------
// Global attention via WMMA (flash-style): block per (b,h,v) group.
// ---------------------------------------------------------------------------
__global__ __launch_bounds__(256) void global_attn_wmma(const _Float16* __restrict__ Q,
                                                        const _Float16* __restrict__ KVG,
                                                        _Float16* __restrict__ OUT) {
  __shared__ _Float16 sKT[64][56];     // sKT[d][j]  (j < 48, zero for j >= 42)
  __shared__ _Float16 sV[64][72];      // sV[j][d]   (zero rows j >= 42)
  __shared__ _Float16 sP[8][16][80];   // per-wave P tile, [row][col], 64 cols used
  int blk = blockIdx.x;                // 0..3199 = (b,h,v)
  int v = blk % V_;
  int r = blk / V_;
  int h = r & 7;
  int b = r >> 3;
  int tid  = threadIdx.x;
  int lane = tid & 31;
  int w    = tid >> 5;

  for (int idx = tid; idx < 48 * 64; idx += 256) {
    int j = idx >> 6;                  // 0..47
    int d = idx & 63;
    _Float16 kk = (_Float16)0.f, vv = (_Float16)0.f;
    if (j < NT_) {
      size_t kvtok = (size_t)(b * NT_ + j) * V_ + v;
      kk = KVG[kvtok * 1024 + h * 64 + d];
      vv = KVG[kvtok * 1024 + 512 + h * 64 + d];
    }
    sKT[d][j] = kk;
    sV[j][d]  = vv;
  }
  for (int idx = tid; idx < 16 * 64; idx += 256) {   // zero V rows 48..63
    sV[48 + (idx >> 6)][idx & 63] = (_Float16)0.f;
  }
  __syncthreads();

  for (int mt = w; mt < 19; mt += 8) {               // ceil(294/16) = 19 tiles
    int m0 = mt * 16;
    int qm = m0 + (lane & 15);
    int kb = (lane >> 4) * 8;
    bool mOK = (qm < T_);
    const _Float16* qp0 = Q;
    if (mOK) qp0 = Q + (((size_t)b * T_ + qm) * V_ + v) * INNER + h * 64;

    v16h qf[2];
    #pragma unroll
    for (int ks = 0; ks < 2; ++ks) {
      if (mOK) {
        const _Float16* qp = qp0 + ks * 32 + kb;
        #pragma unroll
        for (int e = 0; e < 8; ++e) qf[ks][e] = qp[e];
        #pragma unroll
        for (int e = 0; e < 8; ++e) qf[ks][8 + e] = qp[16 + e];
      } else {
        #pragma unroll
        for (int e = 0; e < 16; ++e) qf[ks][e] = (_Float16)0.f;
      }
    }

    // S = Q K^T : 3 N-tiles of 16 kv columns
    v8f sacc[3];
    #pragma unroll
    for (int nt = 0; nt < 3; ++nt) sacc[nt] = v8f{};
    #pragma unroll
    for (int nt = 0; nt < 3; ++nt) {
      #pragma unroll
      for (int ks = 0; ks < 2; ++ks) {
        v16h bf;
        #pragma unroll
        for (int e = 0; e < 16; ++e) bf[e] = sKT[ks * 32 + kb + e + (e & 8)][nt * 16 + (lane & 15)];
        sacc[nt] = wmma_f16(qf[ks], bf, sacc[nt]);
      }
    }

    // row softmax
    bool m2ok = ((lane & 15) + 32) < NT_;            // mask cols >= 42
    int rowBase = (lane >> 4) << 3;
    int col = lane & 15;
    #pragma unroll
    for (int rr = 0; rr < 8; ++rr) {
      float v0 = sacc[0][rr] * 0.125f;
      float v1 = sacc[1][rr] * 0.125f;
      float v2 = m2ok ? sacc[2][rr] * 0.125f : -1.0e30f;
      float mx = fmaxf(fmaxf(v0, v1), v2);
      #pragma unroll
      for (int mk = 8; mk > 0; mk >>= 1) mx = fmaxf(mx, __shfl_xor(mx, mk, 32));
      float e0 = __expf(v0 - mx);
      float e1 = __expf(v1 - mx);
      float e2 = __expf(v2 - mx);
      float ssum = e0 + e1 + e2;
      #pragma unroll
      for (int mk = 8; mk > 0; mk >>= 1) ssum += __shfl_xor(ssum, mk, 32);
      float inv = 1.0f / ssum;
      int row = rowBase + rr;
      sP[w][row][col]      = (_Float16)(e0 * inv);
      sP[w][row][col + 16] = (_Float16)(e1 * inv);
      sP[w][row][col + 32] = (_Float16)(e2 * inv);
      sP[w][row][col + 48] = (_Float16)0.f;
    }

    v16h pf[2];
    #pragma unroll
    for (int ks = 0; ks < 2; ++ks) {
      int pm  = lane & 15;
      int pkb = (lane >> 4) * 8 + ks * 32;
      #pragma unroll
      for (int e = 0; e < 8; ++e) pf[ks][e] = sP[w][pm][pkb + e];
      #pragma unroll
      for (int e = 0; e < 8; ++e) pf[ks][8 + e] = sP[w][pm][pkb + 16 + e];
    }

    // O = P V : 4 N-tiles over d
    v8f oacc[4];
    #pragma unroll
    for (int nt = 0; nt < 4; ++nt) oacc[nt] = v8f{};
    #pragma unroll
    for (int nt = 0; nt < 4; ++nt) {
      #pragma unroll
      for (int ks = 0; ks < 2; ++ks) {
        v16h bf;
        #pragma unroll
        for (int e = 0; e < 16; ++e) bf[e] = sV[ks * 32 + kb + e + (e & 8)][nt * 16 + (lane & 15)];
        oacc[nt] = wmma_f16(pf[ks], bf, oacc[nt]);
      }
    }

    #pragma unroll
    for (int nt = 0; nt < 4; ++nt) {
      #pragma unroll
      for (int rr = 0; rr < 8; ++rr) {
        int gm = m0 + rowBase + rr;
        if (gm < T_) {
          size_t tok = ((size_t)b * T_ + gm) * V_ + v;
          OUT[tok * INNER + h * 64 + nt * 16 + col] = (_Float16)oacc[nt][rr];
        }
      }
    }
  }
}

// ---------------------------------------------------------------------------
// Orchestration
// ---------------------------------------------------------------------------
extern "C" void kernel_launch(void* const* d_in, const int* in_sizes, int n_in,
                              void* d_out, int out_size, void* d_ws, size_t ws_size,
                              hipStream_t stream) {
  (void)in_sizes; (void)n_in; (void)out_size; (void)ws_size;

  const float* x      = (const float*)d_in[0];
  const float* ln1g   = (const float*)d_in[1];
  const float* ln1b   = (const float*)d_in[2];
  const float* la_wq  = (const float*)d_in[3];
  const float* la_wkv = (const float*)d_in[4];
  const float* la_wo  = (const float*)d_in[5];
  const float* la_bo  = (const float*)d_in[6];
  const float* ln2g   = (const float*)d_in[7];
  const float* ln2b   = (const float*)d_in[8];
  const float* f1w1   = (const float*)d_in[9];
  const float* f1b1   = (const float*)d_in[10];
  const float* f1w2   = (const float*)d_in[11];
  const float* f1b2   = (const float*)d_in[12];
  const float* ln3g   = (const float*)d_in[13];
  const float* ln3b   = (const float*)d_in[14];
  const float* ga_wq  = (const float*)d_in[15];
  const float* ga_wkv = (const float*)d_in[16];
  const float* ga_wo  = (const float*)d_in[17];
  const float* ga_bo  = (const float*)d_in[18];
  const float* ln4g   = (const float*)d_in[19];
  const float* ln4b   = (const float*)d_in[20];
  const float* f2w1   = (const float*)d_in[21];
  const float* f2b1   = (const float*)d_in[22];
  const float* f2w2   = (const float*)d_in[23];
  const float* f2b2   = (const float*)d_in[24];

  char* ws = (char*)d_ws;
  const size_t OFF_X   = 0;                                  // f32 [M,128]
  const size_t OFF_LNB = OFF_X   + (size_t)M_ * 128 * 4;     // f16 [M,128]
  const size_t OFF_Q   = OFF_LNB + (size_t)M_ * 128 * 2;     // f16 [M,512] (also FF hidden)
  const size_t OFF_ATT = OFF_Q   + (size_t)M_ * 512 * 2;     // f16 [M,512]
  const size_t OFF_KV  = OFF_ATT + (size_t)M_ * 512 * 2;     // f16 [M,1024] (local kv)
  const size_t OFF_AKV = OFF_KV;                             // f16 [MK,896] (aliased)
  const size_t OFF_KVG = OFF_KV + (size_t)MK_ * KKV * 2;     // f16 [MK,1024] (aliased)

  float*     X    = (float*)(ws + OFF_X);
  _Float16*  LNB  = (_Float16*)(ws + OFF_LNB);
  _Float16*  Qb   = (_Float16*)(ws + OFF_Q);
  _Float16*  ATTb = (_Float16*)(ws + OFF_ATT);
  _Float16*  KVb  = (_Float16*)(ws + OFF_KV);
  _Float16*  AKV  = (_Float16*)(ws + OFF_AKV);
  _Float16*  KVG  = (_Float16*)(ws + OFF_KVG);

  const int ELT_BLKS = (M_ * 128 + 255) / 256;   // 58800
  const int LN_BLKS  = (M_ + 7) / 8;             // 14700
  const dim3 G512(4, (M_ + 127) / 128);          // N=512 GEMMs
  const dim3 G1024(8, (M_ + 127) / 128);         // N=1024 GEMM
  const dim3 G128(1, (M_ + 127) / 128);          // N=128 GEMMs
  const dim3 GKV(8, (MK_ + 127) / 128);          // global kv GEMM
  const int LA_BLKS = (B_ * NT_ * V_ * 8) / 8;   // 16800 blocks, 1 wave/window-head

  to_tokens<<<ELT_BLKS, 256, 0, stream>>>(x, X);

  for (int i = 0; i < 2; ++i) {
    // ---- local attention sublayer ----
    ln_kernel<<<LN_BLKS, 256, 0, stream>>>(X, ln1g + i * 128, ln1b + i * 128, LNB, M_);
    gemm_wmma<0><<<G512, 256, 0, stream>>>(LNB, la_wq + (size_t)i * 128 * 512, nullptr,
                                           Qb, nullptr, M_, 128, 512);
    gemm_wmma<0><<<G1024, 256, 0, stream>>>(LNB, la_wkv + (size_t)i * 128 * 1024, nullptr,
                                            KVb, nullptr, M_, 128, 1024);
    local_attn_wmma<<<LA_BLKS, 256, 0, stream>>>(Qb, KVb, ATTb);
    gemm_wmma<3><<<G128, 256, 0, stream>>>(ATTb, la_wo + (size_t)i * 512 * 128,
                                           la_bo + i * 128, nullptr, X, M_, 512, 128);

    // ---- FF1 sublayer ----
    ln_kernel<<<LN_BLKS, 256, 0, stream>>>(X, ln2g + i * 128, ln2b + i * 128, LNB, M_);
    gemm_wmma<2><<<G512, 256, 0, stream>>>(LNB, f1w1 + (size_t)i * 128 * 512,
                                           f1b1 + i * 512, Qb, nullptr, M_, 128, 512);
    gemm_wmma<3><<<G128, 256, 0, stream>>>(Qb, f1w2 + (size_t)i * 512 * 128,
                                           f1b2 + i * 128, nullptr, X, M_, 512, 128);

    // ---- global attention sublayer ----
    ln_kernel<<<LN_BLKS, 256, 0, stream>>>(X, ln3g + i * 128, ln3b + i * 128, LNB, M_);
    gemm_wmma<0><<<G512, 256, 0, stream>>>(LNB, ga_wq + (size_t)i * 128 * 512, nullptr,
                                           Qb, nullptr, M_, 128, 512);
    akv_gather<<<(MK_ * KKV + 255) / 256, 256, 0, stream>>>(LNB, AKV);
    gemm_wmma<0><<<GKV, 256, 0, stream>>>(AKV, ga_wkv + (size_t)i * KKV * 1024, nullptr,
                                          KVG, nullptr, MK_, KKV, 1024);
    global_attn_wmma<<<B_ * 8 * V_, 256, 0, stream>>>(Qb, KVG, ATTb);
    gemm_wmma<3><<<G128, 256, 0, stream>>>(ATTb, ga_wo + (size_t)i * 512 * 128,
                                           ga_bo + i * 128, nullptr, X, M_, 512, 128);

    // ---- FF2 sublayer ----
    ln_kernel<<<LN_BLKS, 256, 0, stream>>>(X, ln4g + i * 128, ln4b + i * 128, LNB, M_);
    gemm_wmma<2><<<G512, 256, 0, stream>>>(LNB, f2w1 + (size_t)i * 128 * 512,
                                           f2b1 + i * 512, Qb, nullptr, M_, 128, 512);
    gemm_wmma<3><<<G128, 256, 0, stream>>>(Qb, f2w2 + (size_t)i * 512 * 128,
                                           f2b2 + i * 128, nullptr, X, M_, 512, 128);
  }

  from_tokens<<<ELT_BLKS, 256, 0, stream>>>(X, (float*)d_out);
}
